// SSNModel_33646773797630
// MI455X (gfx1250) — compile-verified, matrix operand order checked by reading, not run on hardware
//
#include <hip/hip_runtime.h>
#include <hip/hip_bf16.h>

typedef __attribute__((ext_vector_type(16))) _Float16 v16h;
typedef __attribute__((ext_vector_type(8)))  float    v8f;

#define NPIX   65536
#define NSPIX  512
#define KDIM   144     // 128 feat + 2 coord + 1 ones + 13 zero pad (mult of 16 for N tiles)
#define CFEAT  128
#define EPSF   1e-16f

// ---------------------------------------------------------------------------
// Per-lane fragment pack for v_wmma_f32_16x16x32_f16.
// Row-major operand row base must be pre-offset by (lane>>4)*8 halves.
// Lane holds K = kb..kb+7 and kb+16..kb+23 (ISA 7.12.2 16-bit A/B layout).
// ---------------------------------------------------------------------------
__device__ __forceinline__ v16h pack16(const _Float16* p) {
  v16h r;
#pragma unroll
  for (int i = 0; i < 8; ++i) { r[i] = p[i]; r[i + 8] = p[i + 16]; }
  return r;
}

// ---------------------------------------------------------------------------
// Generic C(MxN,f32) = A(MxK,f16,row-major) * B, with B given as Bt(NxK,f16).
// One 16x16 tile per wave, f32 accumulate in the WMMA.
// ---------------------------------------------------------------------------
__global__ __launch_bounds__(256)
void gemm_f16wmma(const _Float16* __restrict__ A, int lda,
                  const _Float16* __restrict__ Bt, int ldb,
                  float* __restrict__ C, int ldc,
                  int M, int N, int K) {
  int wave = (blockIdx.x * 256 + threadIdx.x) >> 5;
  int lane = threadIdx.x & 31;
  int nT = N >> 4;
  int mTile = wave / nT;
  int nTile = wave - mTile * nT;
  if (mTile >= (M >> 4)) return;
  int hf = lane >> 4;      // 0/1: which K-phase this lane feeds
  int rc = lane & 15;      // row of A / row of Bt (col of B)
  const _Float16* ar = A  + (size_t)(mTile * 16 + rc) * lda + hf * 8;
  const _Float16* br = Bt + (size_t)(nTile * 16 + rc) * ldb + hf * 8;
  v8f acc = {};
  for (int k = 0; k < K; k += 32) {
    v16h av = pack16(ar + k);
    v16h bv = pack16(br + k);
    acc = __builtin_amdgcn_wmma_f32_16x16x32_f16(false, av, false, bv,
                                                 (short)0, acc, false, false);
  }
  float* cr = C + (size_t)(mTile * 16 + hf * 8) * ldc + nTile * 16 + rc;
#pragma unroll
  for (int r = 0; r < 8; ++r) cr[(size_t)r * ldc] = acc[r];
}

// ---------------------------------------------------------------------------
// Fused dist -> softmax(-dist*temp, axis=spix) kernel.
// Block = 256 thr (8 waves) handles 16 pixels x all 512 superpixels.
// Feature dot via WMMA (K=128); coordinate part (K=2) in f32 scalar.
// mode 0: write aff f16 (for the center-update GEMM).
// mode 1: write abs_affinity f32, affinity_matrix=exp(-dist*temp), argmax labels.
// ---------------------------------------------------------------------------
__global__ __launch_bounds__(256)
void dist_softmax(const _Float16* __restrict__ xT,   // NPIX x KDIM
                  const _Float16* __restrict__ ch,   // NSPIX x KDIM
                  const float* __restrict__ centers, // NSPIX x KDIM (f32)
                  const float* __restrict__ cf2, const float* __restrict__ cp2,
                  const float* __restrict__ cs,  const float* __restrict__ psc,
                  const float* __restrict__ f2,  const float* __restrict__ p2,
                  const float* __restrict__ coord,   // 2 x NPIX
                  const float* __restrict__ tempP,
                  _Float16* affh, float* absAff, float* affMat, float* labels,
                  int mode) {
  __shared__ float sD[NSPIX * 16];
  __shared__ float sCs[NSPIX], sPs[NSPIX], sCf2[NSPIX], sCp2[NSPIX];
  __shared__ float sCy[NSPIX], sCx[NSPIX];
  __shared__ float sF2[16], sP2[16], sPy[16], sPx[16];
  __shared__ float sRed[256];
  __shared__ int   sRedI[256];
  __shared__ float sMax[16], sSum[16];
  __shared__ int   sArg[16];

  int t = threadIdx.x;
  int p0 = blockIdx.x * 16;
  float temp = tempP[0];

  for (int i = t; i < NSPIX; i += 256) {
    sCs[i] = cs[i]; sPs[i] = psc[i]; sCf2[i] = cf2[i]; sCp2[i] = cp2[i];
    sCy[i] = centers[i * KDIM + 128]; sCx[i] = centers[i * KDIM + 129];
  }
  if (t < 16) {
    int p = p0 + t;
    sF2[t] = f2[p]; sP2[t] = p2[p];
    sPy[t] = coord[p]; sPx[t] = coord[NPIX + p];
  }
  __syncthreads();

  int wave = t >> 5, lane = t & 31;
  int hf = lane >> 4, rc = lane & 15;
  const _Float16* br = xT + (size_t)(p0 + rc) * KDIM + hf * 8;
#pragma unroll
  for (int tl = 0; tl < 4; ++tl) {
    int mBase = wave * 64 + tl * 16;
    const _Float16* ar = ch + (size_t)(mBase + rc) * KDIM + hf * 8;
    v8f acc = {};
#pragma unroll
    for (int k = 0; k < CFEAT; k += 32) {
      v16h av = pack16(ar + k);
      v16h bv = pack16(br + k);
      acc = __builtin_amdgcn_wmma_f32_16x16x32_f16(false, av, false, bv,
                                                   (short)0, acc, false, false);
    }
#pragma unroll
    for (int r = 0; r < 8; ++r) {
      int s = mBase + hf * 8 + r;
      int n = rc;
      float dotf = acc[r];
      float dotp = sCy[s] * sPy[n] + sCx[s] * sPx[n];
      float df = sCf2[s] + sF2[n] - 2.0f * dotf;
      float dp = sCp2[s] + sP2[n] - 2.0f * dotp;
      sD[s * 16 + n] = sCs[s] * df + sPs[s] * dp;
    }
  }
  __syncthreads();

  // softmax over s: 16 threads per pixel, 32 s-values each (ascending order
  // for first-max argmax semantics)
  int pix = t & 15, chunk = t >> 4;
  float m = -3.4e38f; int arg = 0;
  for (int j = 0; j < 32; ++j) {
    int s = chunk * 32 + j;
    float v = -sD[s * 16 + pix] * temp;
    if (v > m) { m = v; arg = s; }
  }
  sRed[t] = m; sRedI[t] = arg;
  __syncthreads();
  if (chunk == 0) {
    float bm = sRed[pix]; int bi = sRedI[pix];
    for (int j = 1; j < 16; ++j) {
      float v = sRed[j * 16 + pix];
      if (v > bm) { bm = v; bi = sRedI[j * 16 + pix]; }
    }
    sMax[pix] = bm; sArg[pix] = bi;
  }
  __syncthreads();
  float mx = sMax[pix], psum = 0.0f;
  for (int j = 0; j < 32; ++j) {
    int s = chunk * 32 + j;
    float e = __expf(-sD[s * 16 + pix] * temp - mx);
    sD[s * 16 + pix] = e;   // each thread overwrites only its own entries
    psum += e;
  }
  sRed[t] = psum;
  __syncthreads();
  if (chunk == 0) {
    float sm = 0.0f;
    for (int j = 0; j < 16; ++j) sm += sRed[j * 16 + pix];
    sSum[pix] = sm;
  }
  __syncthreads();
  float inv = 1.0f / sSum[pix];
  float sc = __expf(mx);    // exp(-dist*temp) = e * exp(max)
  int p = p0 + pix;
  for (int j = 0; j < 32; ++j) {
    int s = chunk * 32 + j;
    float e = sD[s * 16 + pix];
    size_t idx = (size_t)s * NPIX + p;
    if (mode == 0) {
      affh[idx] = (_Float16)(e * inv);
    } else {
      absAff[idx] = e * inv;
      affMat[idx] = e * sc;
    }
  }
  if (mode == 1 && t < 16) labels[p0 + t] = (float)sArg[t];
}

// ---------------------------------------------------------------------------
// Small helpers
// ---------------------------------------------------------------------------
__global__ void prep_scales(const float* a, float* cs, float* ps) {
  int s = blockIdx.x * blockDim.x + threadIdx.x;
  if (s >= NSPIX) return;
  float sig = 1.0f / (1.0f + __expf(-a[s]));
  float c = sig * 0.8f + 0.1f;
  cs[s] = c; ps[s] = 1.0f - c;
}

__global__ void prep_x(const float* __restrict__ Feature, const float* __restrict__ noise,
                       const float* __restrict__ coord,
                       _Float16* __restrict__ xT, _Float16* __restrict__ xdim,
                       float* __restrict__ f2, float* __restrict__ p2) {
  int p = blockIdx.x * blockDim.x + threadIdx.x;
  if (p >= NPIX) return;
  float s2 = 0.0f;
  for (int d = 0; d < CFEAT; ++d) {
    float v = Feature[(size_t)d * NPIX + p] + noise[(size_t)d * NPIX + p];
    s2 += v * v;
    _Float16 h = (_Float16)v;
    xT[(size_t)p * KDIM + d] = h;
    xdim[(size_t)d * NPIX + p] = h;
  }
  float y = coord[p], x = coord[NPIX + p];
  xT[(size_t)p * KDIM + 128] = (_Float16)y;
  xT[(size_t)p * KDIM + 129] = (_Float16)x;
  xT[(size_t)p * KDIM + 130] = (_Float16)1.0f;
  for (int d = 131; d < KDIM; ++d) xT[(size_t)p * KDIM + d] = (_Float16)0.0f;
  xdim[(size_t)128 * NPIX + p] = (_Float16)y;
  xdim[(size_t)129 * NPIX + p] = (_Float16)x;
  xdim[(size_t)130 * NPIX + p] = (_Float16)1.0f;
  for (int d = 131; d < KDIM; ++d) xdim[(size_t)d * NPIX + p] = (_Float16)0.0f;
  f2[p] = s2; p2[p] = y * y + x * x;
}

// init_labels = p/128 -> superpixel s owns pixels [128s, 128s+128)
__global__ void init_center_acc(const float* Feature, const float* noise,
                                const float* coord, float* acc) {
  int s = blockIdx.x, d = threadIdx.x;     // 144 threads
  int base = s * 128;
  float sum = 0.0f;
  if (d < CFEAT) {
    for (int i = 0; i < 128; ++i) {
      int p = base + i;
      sum += Feature[(size_t)d * NPIX + p] + noise[(size_t)d * NPIX + p];
    }
  } else if (d < 130) {
    for (int i = 0; i < 128; ++i) sum += coord[(size_t)(d - 128) * NPIX + base + i];
  } else if (d == 130) sum = 128.0f;
  acc[s * KDIM + d] = sum;
}

__global__ void center_finalize(const float* __restrict__ acc, float* __restrict__ centers,
                                _Float16* __restrict__ ch, float* __restrict__ cf2,
                                float* __restrict__ cp2) {
  int s = blockIdx.x, d = threadIdx.x;     // 144 threads
  __shared__ float red[KDIM];
  float w = acc[s * KDIM + 130] + EPSF;
  float c = (d < 130) ? acc[s * KDIM + d] / w : 0.0f;
  centers[s * KDIM + d] = c;
  ch[s * KDIM + d] = (_Float16)c;
  red[d] = (d < CFEAT) ? c * c : 0.0f;
  __syncthreads();
  if (d == 0) {
    float t = 0.0f;
    for (int i = 0; i < CFEAT; ++i) t += red[i];
    cf2[s] = t;
    float cy = centers[s * KDIM + 128], cx = centers[s * KDIM + 129];
    cp2[s] = cy * cy + cx * cx;
  }
}

__global__ void build_xn(const float* __restrict__ centers, float* __restrict__ spf,
                         float* __restrict__ xnOut, _Float16* __restrict__ xnT,
                         _Float16* __restrict__ xnD) {
  int s = blockIdx.x, d = threadIdx.x;     // 128 threads
  __shared__ float red[CFEAT];
  __shared__ float nrm;
  float c = centers[s * KDIM + d];
  red[d] = c * c;
  __syncthreads();
  if (d == 0) {
    float t = 0.0f;
    for (int i = 0; i < CFEAT; ++i) t += red[i];
    nrm = sqrtf(t) + EPSF;
  }
  __syncthreads();
  float xn = c / nrm;
  xnOut[(size_t)d * NSPIX + s] = xn;
  xnT[(size_t)s * CFEAT + d] = (_Float16)xn;
  xnD[(size_t)d * NSPIX + s] = (_Float16)xn;
  spf[(size_t)d * NSPIX + s] = c;
  if (d < 2) spf[(size_t)(CFEAT + d) * NSPIX + s] = centers[s * KDIM + 128 + d];
}

__global__ void make_M(const float* gram, float* Mm, _Float16* Mh) {
  int i = blockIdx.x * blockDim.x + threadIdx.x;
  if (i >= NSPIX * NSPIX) return;
  float v = gram[i] + (((i / NSPIX) == (i % NSPIX)) ? 1.0f : 0.0f);
  Mm[i] = v; Mh[i] = (_Float16)v;
}

__global__ void row_abs_sum(const float* Mm, float* rs) {
  int r = blockIdx.x, t = threadIdx.x;     // 128 threads
  __shared__ float red[128];
  float s = 0.0f;
  for (int c = t; c < NSPIX; c += 128) s += fabsf(Mm[r * NSPIX + c]);
  red[t] = s; __syncthreads();
  for (int o = 64; o > 0; o >>= 1) { if (t < o) red[t] += red[t + o]; __syncthreads(); }
  if (t == 0) rs[r] = red[0];
}

__global__ void max_invsq(const float* rs, float* scal) {
  __shared__ float red[NSPIX];
  int t = threadIdx.x;
  red[t] = rs[t]; __syncthreads();
  for (int o = 256; o > 0; o >>= 1) { if (t < o) red[t] = fmaxf(red[t], red[t + o]); __syncthreads(); }
  if (t == 0) scal[0] = 1.0f / (red[0] * red[0]);
}

__global__ void scale_init(const float* Mm, const float* scal, float* X, _Float16* Xh) {
  int i = blockIdx.x * blockDim.x + threadIdx.x;
  if (i >= NSPIX * NSPIX) return;
  float v = Mm[i] * scal[0];
  X[i] = v; Xh[i] = (_Float16)v;
}

__global__ void cvt_f16(const float* in, _Float16* out, int n) {
  int i = blockIdx.x * blockDim.x + threadIdx.x;
  if (i < n) out[i] = (_Float16)in[i];
}

__global__ void ns_update(float* X, const float* Q, _Float16* Xh, int n) {
  int i = blockIdx.x * blockDim.x + threadIdx.x;
  if (i >= n) return;
  float v = 2.0f * X[i] - Q[i];
  X[i] = v; Xh[i] = (_Float16)v;
}

__global__ void zero_f32(float* p, int n) {
  int i = blockIdx.x * blockDim.x + threadIdx.x;
  if (i < n) p[i] = 0.0f;
}

__global__ void copy_f32(const float* in, float* out, int n) {
  int i = blockIdx.x * blockDim.x + threadIdx.x;
  if (i < n) out[i] = in[i];
}

__global__ void transpose_f16(const float* in, _Float16* out, int R, int C) {
  int i = blockIdx.x * blockDim.x + threadIdx.x;
  if (i >= R * C) return;
  int r = i / C, c = i - r * C;
  out[(size_t)c * R + r] = (_Float16)in[i];
}

// Rt = (A + Z - U)^T in f16   (RHO = 1)
__global__ void admm_rt(const float* gram, const float* Zm, const float* Um, _Float16* Rt) {
  int i = blockIdx.x * blockDim.x + threadIdx.x;
  if (i >= NSPIX * NSPIX) return;
  int k = i / NSPIX, n = i - k * NSPIX;
  Rt[(size_t)n * NSPIX + k] = (_Float16)(gram[i] + Zm[i] - Um[i]);
}

__global__ void admm_update(const float* Csol, float* Zm, float* Um,
                            const float* lamda, int l) {
  int i = blockIdx.x * blockDim.x + threadIdx.x;
  if (i >= NSPIX * NSPIX) return;
  float thr = __expf(lamda[l]);            // RHO = 1
  float V = Csol[i] + Um[i];
  float z = copysignf(fmaxf(fabsf(V) - thr, 0.0f), V);
  if ((i / NSPIX) == (i % NSPIX)) z = 0.0f;
  Um[i] = Um[i] + Csol[i] - z;
  Zm[i] = z;
}

__global__ void recon_norm(const float* recon, float* out) {
  int s = blockIdx.x, d = threadIdx.x;     // 128 threads
  __shared__ float red[CFEAT];
  __shared__ float nrm;
  float v = recon[(size_t)d * NSPIX + s];
  red[d] = v * v; __syncthreads();
  if (d == 0) {
    float t = 0.0f;
    for (int i = 0; i < CFEAT; ++i) t += red[i];
    nrm = sqrtf(t) + EPSF;
  }
  __syncthreads();
  out[(size_t)d * NSPIX + s] = v / nrm;
}

// ---------------------------------------------------------------------------
extern "C" void kernel_launch(void* const* d_in, const int* in_sizes, int n_in,
                              void* d_out, int out_size, void* d_ws, size_t ws_size,
                              hipStream_t stream) {
  (void)in_sizes; (void)n_in; (void)out_size; (void)ws_size;
  const float* Feature = (const float*)d_in[0];
  const float* coord   = (const float*)d_in[1];
  const float* a_in    = (const float*)d_in[2];
  const float* noise   = (const float*)d_in[3];
  const float* lamda   = (const float*)d_in[4];
  const float* temp    = (const float*)d_in[5];
  float* out = (float*)d_out;

  // output offsets (flat f32, tuple return order)
  size_t offAbs  = 0;
  size_t offLab  = (size_t)NSPIX * NPIX;
  size_t offSpf  = offLab + NPIX;
  size_t offRec  = offSpf + 130 * NSPIX;
  size_t offCsol = offRec + (size_t)CFEAT * NSPIX;
  size_t offZ    = offCsol + (size_t)NSPIX * NSPIX;
  size_t offXn   = offZ + (size_t)NSPIX * NSPIX;
  size_t offAffM = offXn + (size_t)CFEAT * NSPIX;

  // workspace bump allocator
  char* w = (char*)d_ws;
  auto alloc = [&](size_t b) -> char* {
    char* p = w; w += (b + 255) & ~(size_t)255; return p;
  };
  _Float16* xT    = (_Float16*)alloc((size_t)NPIX * KDIM * 2);
  _Float16* xdim  = (_Float16*)alloc((size_t)KDIM * NPIX * 2);
  float*    f2    = (float*)alloc((size_t)NPIX * 4);
  float*    p2    = (float*)alloc((size_t)NPIX * 4);
  float*    cs    = (float*)alloc(NSPIX * 4);
  float*    psc   = (float*)alloc(NSPIX * 4);
  float*    ctr   = (float*)alloc(NSPIX * KDIM * 4);
  _Float16* ch    = (_Float16*)alloc(NSPIX * KDIM * 2);
  float*    cf2   = (float*)alloc(NSPIX * 4);
  float*    cp2   = (float*)alloc(NSPIX * 4);
  float*    Cacc  = (float*)alloc(NSPIX * KDIM * 4);
  float*    gram  = (float*)alloc((size_t)NSPIX * NSPIX * 4);
  float*    Mm    = (float*)alloc((size_t)NSPIX * NSPIX * 4);
  _Float16* Mh    = (_Float16*)alloc((size_t)NSPIX * NSPIX * 2);
  float*    X     = (float*)alloc((size_t)NSPIX * NSPIX * 4);
  _Float16* Xh    = (_Float16*)alloc((size_t)NSPIX * NSPIX * 2);
  float*    P     = (float*)alloc((size_t)NSPIX * NSPIX * 4);
  _Float16* Ph    = (_Float16*)alloc((size_t)NSPIX * NSPIX * 2);
  float*    Q     = (float*)alloc((size_t)NSPIX * NSPIX * 4);
  float*    rsum  = (float*)alloc(NSPIX * 4);
  float*    scal  = (float*)alloc(256);
  float*    Zm    = (float*)alloc((size_t)NSPIX * NSPIX * 4);
  float*    Um    = (float*)alloc((size_t)NSPIX * NSPIX * 4);
  float*    CsolB = (float*)alloc((size_t)NSPIX * NSPIX * 4);
  _Float16* Rt    = (_Float16*)alloc((size_t)NSPIX * NSPIX * 2);
  _Float16* Zt    = (_Float16*)alloc((size_t)NSPIX * NSPIX * 2);
  _Float16* xnT   = (_Float16*)alloc((size_t)NSPIX * CFEAT * 2);
  _Float16* xnD   = (_Float16*)alloc((size_t)CFEAT * NSPIX * 2);
  float*    recon = (float*)alloc((size_t)CFEAT * NSPIX * 4);

  // f16 affinity scratch lives in the abs_affinity output region (134MB f32;
  // only needs 67MB as f16), overwritten by the final pass.
  _Float16* affh = (_Float16*)(out + offAbs);

  const int EW = 1024;  // elementwise grid for 512x512 = 262144 / 256

  // ---- prep ----
  prep_scales<<<2, 256, 0, stream>>>(a_in, cs, psc);
  prep_x<<<NPIX / 256, 256, 0, stream>>>(Feature, noise, coord, xT, xdim, f2, p2);
  init_center_acc<<<NSPIX, KDIM, 0, stream>>>(Feature, noise, coord, Cacc);
  center_finalize<<<NSPIX, KDIM, 0, stream>>>(Cacc, ctr, ch, cf2, cp2);

  // ---- 20 soft-SLIC iterations (all L2-resident) ----
  int updWaves = (NSPIX / 16) * (KDIM / 16);   // 32*9 = 288
  for (int it = 0; it < 20; ++it) {
    dist_softmax<<<NPIX / 16, 256, 0, stream>>>(xT, ch, ctr, cf2, cp2, cs, psc,
                                                f2, p2, coord, temp,
                                                affh, nullptr, nullptr, nullptr, 0);
    gemm_f16wmma<<<(updWaves + 7) / 8, 256, 0, stream>>>(affh, NPIX, xdim, NPIX,
                                                         Cacc, KDIM,
                                                         NSPIX, KDIM, NPIX);
    center_finalize<<<NSPIX, KDIM, 0, stream>>>(Cacc, ctr, ch, cf2, cp2);
  }
  // final affinities / labels
  dist_softmax<<<NPIX / 16, 256, 0, stream>>>(xT, ch, ctr, cf2, cp2, cs, psc,
                                              f2, p2, coord, temp,
                                              affh, out + offAbs, out + offAffM,
                                              out + offLab, 1);

  // ---- Xn, Gram, M = A + I ----
  build_xn<<<NSPIX, CFEAT, 0, stream>>>(ctr, out + offSpf, out + offXn, xnT, xnD);
  {
    int wv = (NSPIX / 16) * (NSPIX / 16);      // 1024
    gemm_f16wmma<<<(wv + 7) / 8, 256, 0, stream>>>(xnT, CFEAT, xnT, CFEAT,
                                                   gram, NSPIX,
                                                   NSPIX, NSPIX, CFEAT);
  }
  make_M<<<EW, 256, 0, stream>>>(gram, Mm, Mh);

  // ---- Newton-Schulz inverse of M (Pan-Reif init; SPD, guaranteed conv.) ----
  row_abs_sum<<<NSPIX, 128, 0, stream>>>(Mm, rsum);
  max_invsq<<<1, NSPIX, 0, stream>>>(rsum, scal);
  scale_init<<<EW, 256, 0, stream>>>(Mm, scal, X, Xh);
  {
    int wv = (NSPIX / 16) * (NSPIX / 16);
    int gb = (wv + 7) / 8;
    for (int i = 0; i < 18; ++i) {
      gemm_f16wmma<<<gb, 256, 0, stream>>>(Mh, NSPIX, Xh, NSPIX, P, NSPIX,
                                           NSPIX, NSPIX, NSPIX);   // P = M*X (sym)
      cvt_f16<<<EW, 256, 0, stream>>>(P, Ph, NSPIX * NSPIX);
      gemm_f16wmma<<<gb, 256, 0, stream>>>(Xh, NSPIX, Ph, NSPIX, Q, NSPIX,
                                           NSPIX, NSPIX, NSPIX);   // Q = X*M*X
      ns_update<<<EW, 256, 0, stream>>>(X, Q, Xh, NSPIX * NSPIX);  // X = 2X - Q
    }
  }

  // ---- ADMM layers (RHO = 1) ----
  zero_f32<<<EW, 256, 0, stream>>>(Zm, NSPIX * NSPIX);
  zero_f32<<<EW, 256, 0, stream>>>(Um, NSPIX * NSPIX);
  {
    int wv = (NSPIX / 16) * (NSPIX / 16);
    int gb = (wv + 7) / 8;
    for (int l = 0; l < 10; ++l) {
      admm_rt<<<EW, 256, 0, stream>>>(gram, Zm, Um, Rt);
      gemm_f16wmma<<<gb, 256, 0, stream>>>(Xh, NSPIX, Rt, NSPIX, CsolB, NSPIX,
                                           NSPIX, NSPIX, NSPIX);   // Csol = Minv*R
      admm_update<<<EW, 256, 0, stream>>>(CsolB, Zm, Um, lamda, l);
    }
  }
  copy_f32<<<EW, 256, 0, stream>>>(CsolB, out + offCsol, NSPIX * NSPIX);
  copy_f32<<<EW, 256, 0, stream>>>(Zm, out + offZ, NSPIX * NSPIX);

  // ---- recon = Xn @ Z, column-normalized ----
  transpose_f16<<<EW, 256, 0, stream>>>(Zm, Zt, NSPIX, NSPIX);
  {
    int wv = (CFEAT / 16) * (NSPIX / 16);      // 256
    gemm_f16wmma<<<(wv + 7) / 8, 256, 0, stream>>>(xnD, NSPIX, Zt, NSPIX,
                                                   recon, NSPIX,
                                                   CFEAT, NSPIX, NSPIX);
  }
  recon_norm<<<NSPIX, CFEAT, 0, stream>>>(recon, out + offRec);
}
